// Decoder_34119220199683
// MI455X (gfx1250) — matrix operand order repaired and checked by reading, not compiled, not run on hardware
//
#include <hip/hip_runtime.h>
#include <hip/hip_bf16.h>

// ---------------------------------------------------------------------------
// Decoder: 300-step attention LSTM (LAS-style) for MI455X / gfx1250 (wave32).
//   * Weights pre-packed to bf16 WMMA B-fragment layout once per launch.
//   * LSTM gate GEMMs via v_wmma_f32_16x16x32_bf16; one wave owns all 4 gate
//     tiles of one (m,u) 16x16 tile -> cell nonlinearity fully in registers.
//     B fragments for the 4 gates use distinct registers so loads overlap
//     WMMA execution (staggered s_wait_loadcnt instead of full drains).
//   * lstm1: 16 WGs (32 units each), lstm2: 4 WGs -> 1 tile-pair per wave.
//   * Attention f32, loops bounded by lens[n] (masked weights are exactly 0),
//     halving expected L2 traffic; key+values (131MB) are L2-resident (192MB).
// ---------------------------------------------------------------------------

typedef unsigned int u32;
typedef __attribute__((ext_vector_type(4)))  u32    v4u;
typedef __attribute__((ext_vector_type(8)))  float  v8f;
typedef __attribute__((ext_vector_type(16))) __bf16 v16bf;

#define DEC_V   35
#define DEC_E   256
#define DEC_H   512
#define DEC_K   128
#define DEC_VS  128
#define DEC_N   64
#define DEC_T   2000
#define DEC_L   300
#define DEC_START 33
#define INPW    (DEC_E + DEC_VS)   // 384, LSTM1 input width

union FragAB { v16bf v; v4u q[2]; };

__device__ __forceinline__ v16bf load_a_frag(const __bf16* base, int stride,
                                             int m0, int k0, int lane) {
  // 16-bit A 16x32: lanes 0-15 row m, K {0..7,16..23}; lanes 16-31 K {8..15,24..31}
  const __bf16* p = base + (size_t)(m0 + (lane & 15)) * stride + k0 + ((lane >> 4) << 3);
  FragAB f;
  f.q[0] = *(const v4u*)p;
  f.q[1] = *(const v4u*)(p + 16);
  return f.v;
}

__device__ __forceinline__ v16bf load_b_frag(const u32* tile, int lane) {
  // pre-packed: per-lane 16B chunk per 4-VGPR half of the 32x16 B tile
  FragAB f;
  f.q[0] = *(const v4u*)(tile + lane * 4);
  f.q[1] = *(const v4u*)(tile + 128 + lane * 4);
  return f.v;
}

__device__ __forceinline__ v8f wmma_bf16(v16bf a, v16bf b, v8f c) {
  return __builtin_amdgcn_wmma_f32_16x16x32_bf16(false, a, false, b, (short)0, c,
                                                 false, false);
}

__device__ __forceinline__ float sigmoidf(float x) {
  return 1.0f / (1.0f + __expf(-x));
}

__device__ __forceinline__ u32 pack_bf16x2(float lo, float hi) {
  union { __bf16 h[2]; u32 u; } x;
  x.h[0] = (__bf16)lo; x.h[1] = (__bf16)hi;
  return x.u;
}

// ---------------------------------------------------------------------------
// Prep: pack f32 weight W[U x Kd] (row-major) into WMMA B-fragment tiles.
// Tile (ut,kt) = 256 dwords; tix = ut*(Kd/32)+kt. Rows >= Ureal -> 0 pad.
// ---------------------------------------------------------------------------
__global__ __launch_bounds__(256) void pack_b_kernel(const float* __restrict__ W,
                                                     u32* __restrict__ pack,
                                                     int U, int Ureal, int Kd) {
  int i = blockIdx.x * 256 + threadIdx.x;          // dword index
  int total = (U * Kd) >> 1;
  if (i >= total) return;
  int KT   = Kd >> 5;
  int tile = i >> 8;
  int w    = i & 255;
  int ut = tile / KT, kt = tile % KT;
  int c = w >> 7;
  int n = (w >> 2) & 31;
  int j = c * 4 + (w & 3);
  int u = ut * 16 + (n & 15);
  int k = kt * 32 + ((n >> 4) << 3) + ((j >> 2) << 4) + 2 * (j & 3);
  float lo = 0.f, hi = 0.f;
  if (u < Ureal) {
    lo = W[(size_t)u * Kd + k];
    hi = W[(size_t)u * Kd + k + 1];
  }
  pack[i] = pack_bf16x2(lo, hi);
}

// Prep: teacher-forced char embeddings -> inp_all[l][n][0:256] (bf16);
// ctx columns of step 0 zeroed (later steps written by attn of step l-1).
__global__ __launch_bounds__(256) void embed_kernel(const int* __restrict__ text,
                                                    const float* __restrict__ emb,
                                                    __bf16* __restrict__ inp_all) {
  size_t i = (size_t)blockIdx.x * 256 + threadIdx.x;
  if (i >= (size_t)DEC_L * DEC_N * INPW) return;
  int e = (int)(i % INPW);
  size_t r = i / INPW;
  int n = (int)(r % DEC_N);
  int l = (int)(r / DEC_N);
  if (e < DEC_E) {
    int ch = (l == 0) ? DEC_START : text[n * DEC_L + (l - 1)];
    inp_all[i] = (__bf16)emb[ch * DEC_E + e];
  } else if (l == 0) {
    inp_all[i] = (__bf16)0.f;
  }
}

__global__ __launch_bounds__(256) void zero_kernel(u32* __restrict__ p, int nwords) {
  int i = blockIdx.x * 256 + threadIdx.x;
  if (i < nwords) p[i] = 0u;
}

// Four-gate GEMM micro-step: distinct B registers -> overlapped loads/WMMA.
#define GATE_KSTEP(A, PB, GS, KT, kt)                                          \
  do {                                                                         \
    v16bf _b0 = load_b_frag((PB) + ((size_t)((0 * (GS) + ucol) >> 4) * (KT) + (kt)) * 256, lane); \
    v16bf _b1 = load_b_frag((PB) + ((size_t)((1 * (GS) + ucol) >> 4) * (KT) + (kt)) * 256, lane); \
    v16bf _b2 = load_b_frag((PB) + ((size_t)((2 * (GS) + ucol) >> 4) * (KT) + (kt)) * 256, lane); \
    v16bf _b3 = load_b_frag((PB) + ((size_t)((3 * (GS) + ucol) >> 4) * (KT) + (kt)) * 256, lane); \
    acc0 = wmma_bf16((A), _b0, acc0);                                          \
    acc1 = wmma_bf16((A), _b1, acc1);                                          \
    acc2 = wmma_bf16((A), _b2, acc2);                                          \
    acc3 = wmma_bf16((A), _b3, acc3);                                          \
  } while (0)

// ---------------------------------------------------------------------------
// LSTM1: gates = inp @ W_ih1^T + h1 @ W_hh1^T (+b); cell update in registers.
// Grid: 16 WGs x 256 thr; WG owns 32 hidden units; each wave owns exactly one
// (mt,ut) pair = 4 gate tiles sharing A fragments. 28 k-steps of bf16 WMMA.
// ---------------------------------------------------------------------------
__global__ __launch_bounds__(256) void lstm1_kernel(
    const __bf16* __restrict__ inp_all, const __bf16* __restrict__ h1r,
    __bf16* __restrict__ h1w, float* __restrict__ c1,
    const u32* __restrict__ pWih, const u32* __restrict__ pWhh,
    const float* __restrict__ bih, const float* __restrict__ bhh, int l) {
  const int KT0 = INPW / 32;   // 12
  const int KT1 = DEC_H / 32;  // 16
  const __bf16* A0 = inp_all + (size_t)l * DEC_N * INPW;
  int wid = threadIdx.x >> 5, lane = threadIdx.x & 31;
  int mt = wid >> 1, ut = wid & 1;
  int ucol = blockIdx.x * 32 + ut * 16;
  v8f acc0 = {}, acc1 = {}, acc2 = {}, acc3 = {};
#pragma unroll 2
  for (int kt = 0; kt < KT0; ++kt) {
    v16bf a = load_a_frag(A0, INPW, mt * 16, kt * 32, lane);
    GATE_KSTEP(a, pWih, DEC_H, KT0, kt);
  }
#pragma unroll 2
  for (int kt = 0; kt < KT1; ++kt) {
    v16bf a = load_a_frag(h1r, DEC_H, mt * 16, kt * 32, lane);
    GATE_KSTEP(a, pWhh, DEC_H, KT1, kt);
  }
  int ug = ucol + (lane & 15);
  float b0 = bih[0 * DEC_H + ug] + bhh[0 * DEC_H + ug];
  float b1 = bih[1 * DEC_H + ug] + bhh[1 * DEC_H + ug];
  float b2 = bih[2 * DEC_H + ug] + bhh[2 * DEC_H + ug];
  float b3 = bih[3 * DEC_H + ug] + bhh[3 * DEC_H + ug];
  int mrow = mt * 16 + ((lane >> 4) << 3);
#pragma unroll
  for (int r = 0; r < 8; ++r) {
    size_t ix = (size_t)(mrow + r) * DEC_H + ug;
    float iv = sigmoidf(acc0[r] + b0);
    float fv = sigmoidf(acc1[r] + b1);
    float gv = tanhf(acc2[r] + b2);
    float ov = sigmoidf(acc3[r] + b3);
    float c = fv * c1[ix] + iv * gv;
    c1[ix] = c;
    h1w[ix] = (__bf16)(ov * tanhf(c));
  }
}

// LSTM2: gates = h1 @ W_ih2^T + h2 @ W_hh2^T. Grid: 4 WGs x 256 thr.
__global__ __launch_bounds__(256) void lstm2_kernel(
    const __bf16* __restrict__ h1, const __bf16* __restrict__ h2r,
    __bf16* __restrict__ h2w, float* __restrict__ c2,
    const u32* __restrict__ pWih, const u32* __restrict__ pWhh,
    const float* __restrict__ bih, const float* __restrict__ bhh,
    float* __restrict__ h2f) {
  const int KT0 = DEC_H / 32;  // 16
  const int KT1 = DEC_K / 32;  // 4
  int wid = threadIdx.x >> 5, lane = threadIdx.x & 31;
  int mt = wid >> 1, ut = wid & 1;
  int ucol = blockIdx.x * 32 + ut * 16;
  v8f acc0 = {}, acc1 = {}, acc2 = {}, acc3 = {};
#pragma unroll 2
  for (int kt = 0; kt < KT0; ++kt) {
    v16bf a = load_a_frag(h1, DEC_H, mt * 16, kt * 32, lane);
    GATE_KSTEP(a, pWih, DEC_K, KT0, kt);
  }
#pragma unroll
  for (int kt = 0; kt < KT1; ++kt) {
    v16bf a = load_a_frag(h2r, DEC_K, mt * 16, kt * 32, lane);
    GATE_KSTEP(a, pWhh, DEC_K, KT1, kt);
  }
  int ug = ucol + (lane & 15);
  float b0 = bih[0 * DEC_K + ug] + bhh[0 * DEC_K + ug];
  float b1 = bih[1 * DEC_K + ug] + bhh[1 * DEC_K + ug];
  float b2 = bih[2 * DEC_K + ug] + bhh[2 * DEC_K + ug];
  float b3 = bih[3 * DEC_K + ug] + bhh[3 * DEC_K + ug];
  int mrow = mt * 16 + ((lane >> 4) << 3);
#pragma unroll
  for (int r = 0; r < 8; ++r) {
    size_t ix = (size_t)(mrow + r) * DEC_K + ug;
    float iv = sigmoidf(acc0[r] + b0);
    float fv = sigmoidf(acc1[r] + b1);
    float gv = tanhf(acc2[r] + b2);
    float ov = sigmoidf(acc3[r] + b3);
    float c = fv * c2[ix] + iv * gv;
    c2[ix] = c;
    float h = ov * tanhf(c);
    h2w[ix] = (__bf16)h;
    h2f[ix] = h;
  }
}

// ---------------------------------------------------------------------------
// Attention: per batch element n (64 WGs). energy -> masked softmax -> ctx.
// All loops bounded by lens[n]: masked weights are exactly 0 (exp underflow),
// so skipping t >= len is bit-identical and halves expected L2 traffic.
// ---------------------------------------------------------------------------
__global__ __launch_bounds__(256) void attn_kernel(
    const float* __restrict__ key, const float* __restrict__ values,
    const int* __restrict__ lens, const float* __restrict__ h2f,
    __bf16* __restrict__ inp_all, __bf16* __restrict__ ctxb, int l) {
  __shared__ float sh_h2[DEC_K];
  __shared__ float sh_e[DEC_T];
  __shared__ float red[8];
  __shared__ float sh_ctx[DEC_K];
  int n = blockIdx.x;
  int tid = threadIdx.x, wid = tid >> 5, lane = tid & 31;
  if (tid < DEC_K) sh_h2[tid] = h2f[n * DEC_K + tid];
  __syncthreads();
  int len = lens[n];
  float4 hv = *(const float4*)&sh_h2[lane * 4];
  const float* kb = key + (size_t)n * DEC_T * DEC_K;
  for (int t = wid; t < len; t += 8) {
    const float* kr = kb + (size_t)t * DEC_K + lane * 4;
    __builtin_prefetch(kr + 8 * DEC_K, 0, 0);   // global_prefetch next row
    float4 kv = *(const float4*)kr;
    float d = kv.x * hv.x + kv.y * hv.y + kv.z * hv.z + kv.w * hv.w;
#pragma unroll
    for (int off = 16; off; off >>= 1) d += __shfl_xor(d, off);
    if (lane == 0) sh_e[t] = d;
  }
  __syncthreads();
  // block max over [0, len)
  float m = -3.4e38f;
  for (int t = tid; t < len; t += 256) m = fmaxf(m, sh_e[t]);
#pragma unroll
  for (int off = 16; off; off >>= 1) m = fmaxf(m, __shfl_xor(m, off));
  if (lane == 0) red[wid] = m;
  __syncthreads();
  if (tid == 0) {
    float mm = red[0];
    for (int i = 1; i < 8; ++i) mm = fmaxf(mm, red[i]);
    red[0] = mm;
  }
  __syncthreads();
  float M = red[0];
  __syncthreads();
  // exp + block sum over [0, len)
  float s = 0.f;
  for (int t = tid; t < len; t += 256) {
    float e = __expf(sh_e[t] - M);
    sh_e[t] = e;
    s += e;
  }
#pragma unroll
  for (int off = 16; off; off >>= 1) s += __shfl_xor(s, off);
  if (lane == 0) red[wid] = s;
  __syncthreads();
  if (tid == 0) {
    float ss = 0.f;
    for (int i = 0; i < 8; ++i) ss += red[i];
    red[0] = ss;
  }
  __syncthreads();
  float inv = 1.0f / red[0];
  // ctx = sum_{t<len} attn[t] * values[n,t,:] (two t-halves over 256 threads)
  int k = tid & 127, half = tid >> 7;
  const float* vb = values + (size_t)n * DEC_T * DEC_K;
  int mid = len >> 1;
  int t0 = half ? mid : 0;
  int t1 = half ? len : mid;
  float acc = 0.f;
  for (int t = t0; t < t1; ++t) {
    const float* vp = vb + (size_t)t * DEC_K + k;
    __builtin_prefetch(vp + 16 * DEC_K, 0, 0);
    acc += sh_e[t] * (*vp);
  }
  if (half) sh_ctx[k] = acc;
  __syncthreads();
  if (!half) {
    float c = (acc + sh_ctx[k]) * inv;
    ctxb[n * DEC_K + k] = (__bf16)c;
    if (l + 1 < DEC_L)
      inp_all[((size_t)(l + 1) * DEC_N + n) * INPW + DEC_E + k] = (__bf16)c;
  }
}

// pred = [h2, ctx] @ embedding^T + b_out   (M=64, N=35->48 pad, K=256) via WMMA
__global__ __launch_bounds__(256) void pred_kernel(
    const __bf16* __restrict__ h2b, const __bf16* __restrict__ ctxb,
    const u32* __restrict__ pEmb, const float* __restrict__ bout,
    float* __restrict__ out, int l) {
  int wid = threadIdx.x >> 5, lane = threadIdx.x & 31;
  for (int t = wid; t < 12; t += 8) {
    int mt = t / 3, nt = t - mt * 3;
    v8f acc = {};
#pragma unroll
    for (int kt = 0; kt < 4; ++kt) {   // first 128 cols: h2
      v16bf a = load_a_frag(h2b, DEC_K, mt * 16, kt * 32, lane);
      const u32* bt = pEmb + ((size_t)nt * 8 + kt) * 256;
      acc = wmma_bf16(a, load_b_frag(bt, lane), acc);
    }
#pragma unroll
    for (int kt = 0; kt < 4; ++kt) {   // next 128 cols: ctx
      v16bf a = load_a_frag(ctxb, DEC_K, mt * 16, kt * 32, lane);
      const u32* bt = pEmb + ((size_t)nt * 8 + 4 + kt) * 256;
      acc = wmma_bf16(a, load_b_frag(bt, lane), acc);
    }
    int v = nt * 16 + (lane & 15);
    if (v < DEC_V) {
      float bv = bout[v];
      int mrow = mt * 16 + ((lane >> 4) << 3);
#pragma unroll
      for (int r = 0; r < 8; ++r)
        out[(size_t)(mrow + r) * DEC_L * DEC_V + (size_t)l * DEC_V + v] = acc[r] + bv;
    }
  }
}

// ---------------------------------------------------------------------------
extern "C" void kernel_launch(void* const* d_in, const int* in_sizes, int n_in,
                              void* d_out, int out_size, void* d_ws, size_t ws_size,
                              hipStream_t stream) {
  const float* key    = (const float*)d_in[0];
  const float* values = (const float*)d_in[1];
  const int*   lens   = (const int*)d_in[2];
  const int*   text   = (const int*)d_in[3];
  const float* emb    = (const float*)d_in[4];
  const float* Wih1   = (const float*)d_in[5];
  const float* Whh1   = (const float*)d_in[6];
  const float* bih1   = (const float*)d_in[7];
  const float* bhh1   = (const float*)d_in[8];
  const float* Wih2   = (const float*)d_in[9];
  const float* Whh2   = (const float*)d_in[10];
  const float* bih2   = (const float*)d_in[11];
  const float* bhh2   = (const float*)d_in[12];
  const float* bout   = (const float*)d_in[13];
  float* out = (float*)d_out;
  char*  ws  = (char*)d_ws;

  size_t off = 0;
  auto take = [&](size_t b) { size_t r = off; off += (b + 255) & ~(size_t)255; return r; };

  u32*    pWih1   = (u32*)(ws + take((size_t)4 * DEC_H * INPW * 2));
  u32*    pWhh1   = (u32*)(ws + take((size_t)4 * DEC_H * DEC_H * 2));
  u32*    pWih2   = (u32*)(ws + take((size_t)4 * DEC_K * DEC_H * 2));
  u32*    pWhh2   = (u32*)(ws + take((size_t)4 * DEC_K * DEC_K * 2));
  u32*    pEmb    = (u32*)(ws + take((size_t)48 * DEC_E * 2));
  __bf16* inp_all = (__bf16*)(ws + take((size_t)DEC_L * DEC_N * INPW * 2));
  size_t zero_base = off;                       // contiguous zero region start
  float*  c1   = (float*)(ws + take((size_t)DEC_N * DEC_H * 4));
  float*  c2   = (float*)(ws + take((size_t)DEC_N * DEC_K * 4));
  __bf16* h1b  = (__bf16*)(ws + take((size_t)2 * DEC_N * DEC_H * 2));
  __bf16* h2b  = (__bf16*)(ws + take((size_t)2 * DEC_N * DEC_K * 2));
  size_t zero_words = (off - zero_base) / 4;
  float*  h2f  = (float*)(ws + take((size_t)DEC_N * DEC_K * 4));
  __bf16* ctxb = (__bf16*)(ws + take((size_t)DEC_N * DEC_K * 2));
  (void)in_sizes; (void)n_in; (void)out_size; (void)ws_size;

  // ---- prep (runs every launch; deterministic) ----
  {
    int nd;
    nd = 4 * DEC_H * INPW / 2;
    pack_b_kernel<<<(nd + 255) / 256, 256, 0, stream>>>(Wih1, pWih1, 4 * DEC_H, 4 * DEC_H, INPW);
    nd = 4 * DEC_H * DEC_H / 2;
    pack_b_kernel<<<(nd + 255) / 256, 256, 0, stream>>>(Whh1, pWhh1, 4 * DEC_H, 4 * DEC_H, DEC_H);
    nd = 4 * DEC_K * DEC_H / 2;
    pack_b_kernel<<<(nd + 255) / 256, 256, 0, stream>>>(Wih2, pWih2, 4 * DEC_K, 4 * DEC_K, DEC_H);
    nd = 4 * DEC_K * DEC_K / 2;
    pack_b_kernel<<<(nd + 255) / 256, 256, 0, stream>>>(Whh2, pWhh2, 4 * DEC_K, 4 * DEC_K, DEC_K);
    nd = 48 * DEC_E / 2;
    pack_b_kernel<<<(nd + 255) / 256, 256, 0, stream>>>(emb, pEmb, 48, DEC_V, DEC_E);
    size_t ne = (size_t)DEC_L * DEC_N * INPW;
    embed_kernel<<<(int)((ne + 255) / 256), 256, 0, stream>>>(text, emb, inp_all);
    zero_kernel<<<(int)((zero_words + 255) / 256), 256, 0, stream>>>((u32*)(ws + zero_base),
                                                                     (int)zero_words);
  }

  // ---- 300 sequential decode steps ----
  const size_t H1SZ = (size_t)DEC_N * DEC_H;
  const size_t H2SZ = (size_t)DEC_N * DEC_K;
  for (int l = 0; l < DEC_L; ++l) {
    const __bf16* h1r = h1b + (size_t)(l & 1) * H1SZ;
    __bf16*       h1w = h1b + (size_t)((l + 1) & 1) * H1SZ;
    const __bf16* h2r = h2b + (size_t)(l & 1) * H2SZ;
    __bf16*       h2w = h2b + (size_t)((l + 1) & 1) * H2SZ;
    lstm1_kernel<<<16, 256, 0, stream>>>(inp_all, h1r, h1w, c1, pWih1, pWhh1, bih1, bhh1, l);
    lstm2_kernel<<<4, 256, 0, stream>>>(h1w, h2r, h2w, c2, pWih2, pWhh2, bih2, bhh2, h2f);
    attn_kernel<<<DEC_N, 256, 0, stream>>>(key, values, lens, h2f, inp_all, ctxb, l);
    pred_kernel<<<1, 256, 0, stream>>>(h2w, ctxb, pEmb, bout, out, l);
  }
}
